// EmformerEncoder_41257455845766
// MI455X (gfx1250) — matrix operand order repaired and checked by reading, not compiled, not run on hardware
//
#include <hip/hip_runtime.h>

// ---------------- problem constants (from reference) ----------------
#define CHUNKSZ 32
#define RC      8
#define LC      32
#define DMODEL  256
#define NH      4
#define DH      64          // DMODEL / NH
#define FFDIM   2048
#define NLAYERS 4
#define ULEN    2048
#define BATCH   4
#define TLEN    (ULEN + RC)     // 2056
#define NCHUNK  64              // ceil(U/CHUNK)
#define RLEN    (RC * NCHUNK)   // 512
#define NROWS   (RLEN + ULEN)   // 2560
#define MTOK    (NROWS * BATCH) // 10240 GEMM rows

typedef __attribute__((ext_vector_type(16))) __bf16 v16bf;
typedef __attribute__((ext_vector_type(8)))  float  v8f;

__device__ __forceinline__ unsigned short f2bf(float f) {
    unsigned u = __float_as_uint(f);
    u += 0x7FFFu + ((u >> 16) & 1u);   // round-to-nearest-even
    return (unsigned short)(u >> 16);
}

// ---------------- weight swizzle: f32 [K,N] row-major -> bf16 WMMA-B fragment order -------
// Per 32x16 (KxN) tile, lane's 16 elements contiguous:
//   dst[(((kt*(N/16) + nt)*32 + lane)*16 + e] = W[(kt*32 + (lane>>4)*16 + e)*N + nt*16 + (lane&15)]
__global__ __launch_bounds__(256) void swz_w_kernel(const float* __restrict__ W,
                                                    unsigned short* __restrict__ dst,
                                                    int Ndim, int total) {
    int i = blockIdx.x * 256 + threadIdx.x;
    if (i >= total) return;
    int e    = i & 15;
    int lane = (i >> 4) & 31;
    int tile = i >> 9;
    int ntn  = Ndim >> 4;
    int kt   = tile / ntn;
    int nt   = tile - kt * ntn;
    int k    = kt * 32 + (lane >> 4) * 16 + e;
    int col  = nt * 16 + (lane & 15);
    dst[i] = f2bf(W[(size_t)k * Ndim + col]);
}

// ---------------- gather rc_utt = concat(x[rc_index], x[:U]) ----------------
__global__ __launch_bounds__(256) void gather_kernel(const float* __restrict__ x,
                                                     float* __restrict__ rcu) {
    int i = blockIdx.x * 256 + threadIdx.x;          // over NROWS*BATCH*DMODEL
    if (i >= NROWS * BATCH * DMODEL) return;
    int n  = i / (BATCH * DMODEL);
    int rb = i - n * (BATCH * DMODEL);
    int src;
    if (n < RLEN) {
        int r = n & (RC - 1);
        if (n < RLEN - RC) src = CHUNKSZ * (n >> 3) + CHUNKSZ + r;  // intervals + first
        else               src = (TLEN - RC) + r;                   // tail rows
    } else {
        src = n - RLEN;                                             // out = x[:U]
    }
    rcu[i] = x[(size_t)src * (BATCH * DMODEL) + rb];
}

// ---------------- LayerNorm: one wave32 per token ----------------
__global__ __launch_bounds__(256) void ln_kernel(const float* __restrict__ x,
                                                 const float* __restrict__ w,
                                                 const float* __restrict__ b,
                                                 float* __restrict__ outF,          // optional
                                                 unsigned short* __restrict__ outB, // optional
                                                 int ntok) {
    int wave = (blockIdx.x * 256 + threadIdx.x) >> 5;
    int lane = threadIdx.x & 31;
    if (wave >= ntok) return;
    const float* xp = x + (size_t)wave * DMODEL;
    float v[8];
    float s = 0.f;
#pragma unroll
    for (int i = 0; i < 8; ++i) {
        int c = lane + i * 32;
        float t = xp[c];
        v[i] = t; s += t;
    }
#pragma unroll
    for (int m = 16; m >= 1; m >>= 1) s += __shfl_xor(s, m, 32);
    float mean = s * (1.0f / DMODEL);
    float q = 0.f;
#pragma unroll
    for (int i = 0; i < 8; ++i) { float d = v[i] - mean; q += d * d; }
#pragma unroll
    for (int m = 16; m >= 1; m >>= 1) q += __shfl_xor(q, m, 32);
    float inv = rsqrtf(q * (1.0f / DMODEL) + 1e-5f);
#pragma unroll
    for (int i = 0; i < 8; ++i) {
        int c = lane + i * 32;
        float y = (v[i] - mean) * inv * w[c] + b[c];
        size_t idx = (size_t)wave * DMODEL + c;
        if (outF) outF[idx] = y;
        if (outB) outB[idx] = f2bf(y);
    }
}

// ---------------- WMMA GEMM: C[M,N] = A[M,K](bf16) * W(swizzled bf16) + bias --------------
// One wave per 64x32 output block (4 M-tiles x 2 N-tiles): per K-step load 4 A frags +
// 2 B frags (12x b128), issue 8 v_wmma into independent accumulators (no hazard NOPs,
// partial loadcnt waits overlap the WMMA pipe). A reused 2x, B reused 4x.
// Fragment k order: contiguous, k = 16*(lane/16) + e, same for A and B (wave32).
template<bool RESID, bool OUTF, bool OUTB, bool RELU>
__global__ __launch_bounds__(256) void gemm_bf16_kernel(const unsigned short* __restrict__ A,
                                                        const unsigned short* __restrict__ Wsw,
                                                        const float* __restrict__ bias,
                                                        const float* __restrict__ resid,
                                                        float* __restrict__ outF,
                                                        unsigned short* __restrict__ outB,
                                                        int Mdim, int Ndim, int Kdim) {
    int wave = (blockIdx.x * 256 + threadIdx.x) >> 5;   // wave-uniform
    int lane = threadIdx.x & 31;
    int ntn32 = Ndim >> 5;
    int tiles = (Mdim >> 6) * ntn32;
    if (wave >= tiles) return;                           // uniform per wave: EXEC stays all-ones
    int mt = wave / ntn32, nt = wave - mt * ntn32;
    int m0 = mt << 6, n0 = nt << 5;

    int hi    = lane >> 4;            // half-wave select
    int ahalf = hi ? 16 : 0;          // A: lanes 16-31 carry K 16-31
    int ntn   = Ndim >> 4;

    const unsigned short* Arow[4];
#pragma unroll
    for (int mi = 0; mi < 4; ++mi)
        Arow[mi] = A + (size_t)(m0 + mi * 16 + (lane & 15)) * Kdim;

    v8f acc[4][2];
#pragma unroll
    for (int mi = 0; mi < 4; ++mi)
#pragma unroll
        for (int j = 0; j < 2; ++j)
            acc[mi][j] = (v8f){0.f, 0.f, 0.f, 0.f, 0.f, 0.f, 0.f, 0.f};

    for (int kk = 0; kk < Kdim; kk += 32) {
        if (kk + 32 < Kdim) __builtin_prefetch(Arow[0] + kk + 32, 0, 0);  // global_prefetch_b8
        union Frag { unsigned short u[16]; v16bf v; };
        Frag af[4];
#pragma unroll
        for (int mi = 0; mi < 4; ++mi)
#pragma unroll
            for (int e = 0; e < 16; ++e) {
                int ka = kk + ahalf + e;
                af[mi].u[e] = Arow[mi][ka];
            }
        // swizzled weight tiles: 16 contiguous u16 per lane -> 2x global_load_b128 each
        const unsigned short* Wbase =
            Wsw + (((size_t)(kk >> 5) * ntn + (n0 >> 4)) * 32 + lane) * 16;
        Frag bfr[2];
#pragma unroll
        for (int j = 0; j < 2; ++j)
#pragma unroll
            for (int e = 0; e < 16; ++e) bfr[j].u[e] = Wbase[(size_t)j * 512 + e];
        // 8 back-to-back WMMAs, all-independent destinations
#pragma unroll
        for (int j = 0; j < 2; ++j)
#pragma unroll
            for (int mi = 0; mi < 4; ++mi)
                acc[mi][j] = __builtin_amdgcn_wmma_f32_16x16x32_bf16(
                    false, af[mi].v, false, bfr[j].v, (short)0, acc[mi][j], false, false);
    }
#pragma unroll
    for (int j = 0; j < 2; ++j) {
        int col = n0 + j * 16 + (lane & 15);
        float bb = bias[col];
#pragma unroll
        for (int mi = 0; mi < 4; ++mi)
#pragma unroll
            for (int r = 0; r < 8; ++r) {
                int row = m0 + mi * 16 + r + (hi ? 8 : 0);
                float y = acc[mi][j][r] + bb;
                if (RELU) y = y > 0.f ? y : 0.f;
                size_t idx = (size_t)row * Ndim + col;
                if (RESID) y += resid[idx];
                if (OUTF) outF[idx] = y;
                if (OUTB) outB[idx] = f2bf(y);
            }
    }
}

// ---------------- sparse Emformer attention, online softmax ----------------
// allowed keys per query: 8 rc keys of own chunk + utterance window
// [chunk*32-32, chunk*32+32) clipped to [0, out_len). ~72 keys -> VALU beats dense WMMA.
__global__ __launch_bounds__(256) void attn_kernel(const float* __restrict__ q,
                                                   const float* __restrict__ kv,
                                                   const int* __restrict__ lengths,
                                                   unsigned short* __restrict__ attnB) {
    int t = blockIdx.x * 256 + threadIdx.x;
    if (t >= NROWS * BATCH * NH) return;
    int h  = t & (NH - 1);
    int b  = (t >> 2) & (BATCH - 1);
    int qi = t >> 4;
    int out_len = lengths[b] - RC; if (out_len < 0) out_len = 0;
    int chunk = (qi < RLEN) ? (qi >> 3) : ((qi - RLEN) >> 5);

    float qv[DH];
    const float* qp = q + ((size_t)qi * BATCH + b) * DMODEL + h * DH;
#pragma unroll
    for (int d = 0; d < DH; ++d) qv[d] = qp[d] * 0.125f;   // DH^-0.5

    float m = -3.0e38f, l = 0.f;
    float acc[DH];
#pragma unroll
    for (int d = 0; d < DH; ++d) acc[d] = 0.f;

    auto proc = [&](int kj) {
        const float* kp = kv + ((size_t)kj * BATCH + b) * (2 * DMODEL) + h * DH;
        const float* vp = kp + DMODEL;
        float s = 0.f;
#pragma unroll
        for (int d = 0; d < DH; ++d) s += qv[d] * kp[d];
        float mn = s > m ? s : m;
        float alpha = __expf(m - mn);
        float p = __expf(s - mn);
        l = l * alpha + p;
#pragma unroll
        for (int d = 0; d < DH; ++d) acc[d] = acc[d] * alpha + p * vp[d];
        m = mn;
    };

    for (int kj = chunk * RC; kj < chunk * RC + RC; ++kj) proc(kj);   // rc keys (never pad-masked)
    int lo = chunk * CHUNKSZ - LC; if (lo < 0) lo = 0;
    int hh = (chunk + 1) * CHUNKSZ; if (hh > out_len) hh = out_len;
    for (int tt = lo; tt < hh; ++tt) proc(RLEN + tt);                 // utterance window

    float invl = 1.0f / l;   // l > 0: rc keys always contribute
    unsigned short* op = attnB + ((size_t)qi * BATCH + b) * DMODEL + h * DH;
#pragma unroll
    for (int d = 0; d < DH; ++d) op[d] = f2bf(acc[d] * invl);
}

// ---------------- final: d_out = [ out(U,B,D) f32 , out_lengths(B) ] ----------------
__global__ __launch_bounds__(256) void finalize_kernel(const float* __restrict__ rcu,
                                                       const int* __restrict__ lengths,
                                                       float* __restrict__ out) {
    int i = blockIdx.x * 256 + threadIdx.x;
    const int nmain = ULEN * BATCH * DMODEL;
    if (i < nmain) {
        out[i] = rcu[(size_t)RLEN * BATCH * DMODEL + i];
    } else if (i < nmain + BATCH) {
        int b = i - nmain;
        int ol = lengths[b] - RC; if (ol < 0) ol = 0;
        out[i] = (float)ol;
    }
}

extern "C" void kernel_launch(void* const* d_in, const int* in_sizes, int n_in,
                              void* d_out, int out_size, void* d_ws, size_t ws_size,
                              hipStream_t stream) {
    const float* x        = (const float*)d_in[0];
    const int*   lengths  = (const int*)  d_in[1];
    const float* ln_in_w  = (const float*)d_in[2];
    const float* ln_in_b  = (const float*)d_in[3];
    const float* Wq       = (const float*)d_in[4];
    const float* bq       = (const float*)d_in[5];
    const float* Wkv      = (const float*)d_in[6];
    const float* bkv      = (const float*)d_in[7];
    const float* Wo       = (const float*)d_in[8];
    const float* bo       = (const float*)d_in[9];
    const float* ff_ln_w  = (const float*)d_in[10];
    const float* ff_ln_b  = (const float*)d_in[11];
    const float* W1       = (const float*)d_in[12];
    const float* b1       = (const float*)d_in[13];
    const float* W2       = (const float*)d_in[14];
    const float* b2       = (const float*)d_in[15];
    const float* ln_out_w = (const float*)d_in[16];
    const float* ln_out_b = (const float*)d_in[17];

    char* ws = (char*)d_ws;
    size_t off = 0;
    auto alloc = [&](size_t bytes) -> void* {
        void* p = ws + off;
        off += (bytes + 255) & ~(size_t)255;
        return p;
    };
    const size_t MD = (size_t)MTOK * DMODEL;
    float*          rcu    = (float*)         alloc(MD * 4);
    float*          tmp    = (float*)         alloc(MD * 4);
    float*          qbuf   = (float*)         alloc(MD * 4);
    float*          kvbuf  = (float*)         alloc((size_t)MTOK * 2 * DMODEL * 4);
    unsigned short* h_bf   = (unsigned short*)alloc(MD * 2);
    unsigned short* f_bf   = (unsigned short*)alloc(MD * 2);
    unsigned short* a_bf   = (unsigned short*)alloc(MD * 2);
    unsigned short* ff1_bf = (unsigned short*)alloc((size_t)MTOK * FFDIM * 2);
    unsigned short* wqB    = (unsigned short*)alloc((size_t)NLAYERS * DMODEL * DMODEL * 2);
    unsigned short* wkvB   = (unsigned short*)alloc((size_t)NLAYERS * DMODEL * 2 * DMODEL * 2);
    unsigned short* woB    = (unsigned short*)alloc((size_t)NLAYERS * DMODEL * DMODEL * 2);
    unsigned short* w1B    = (unsigned short*)alloc((size_t)NLAYERS * DMODEL * FFDIM * 2);
    unsigned short* w2B    = (unsigned short*)alloc((size_t)NLAYERS * FFDIM * DMODEL * 2);

    // pre-swizzle all weights into WMMA B-fragment order (per layer slice)
    auto swz = [&](const float* src, unsigned short* dst, int K, int N) {
        int total = K * N;
        for (int l = 0; l < NLAYERS; ++l)
            swz_w_kernel<<<(total + 255) / 256, 256, 0, stream>>>(
                src + (size_t)l * total, dst + (size_t)l * total, N, total);
    };
    swz(Wq,  wqB,  DMODEL, DMODEL);
    swz(Wkv, wkvB, DMODEL, 2 * DMODEL);
    swz(Wo,  woB,  DMODEL, DMODEL);
    swz(W1,  w1B,  DMODEL, FFDIM);
    swz(W2,  w2B,  FFDIM,  DMODEL);

    gather_kernel<<<(NROWS * BATCH * DMODEL + 255) / 256, 256, 0, stream>>>(x, rcu);

    auto blocks = [](int M, int N) { return ((M >> 6) * (N >> 5) + 7) / 8; };
    const int lnBlocks = (MTOK * 32 + 255) / 256;   // one wave per token

    for (int l = 0; l < NLAYERS; ++l) {
        const unsigned short* wq_l  = wqB  + (size_t)l * DMODEL * DMODEL;
        const unsigned short* wkv_l = wkvB + (size_t)l * DMODEL * 2 * DMODEL;
        const unsigned short* wo_l  = woB  + (size_t)l * DMODEL * DMODEL;
        const unsigned short* w1_l  = w1B  + (size_t)l * DMODEL * FFDIM;
        const unsigned short* w2_l  = w2B  + (size_t)l * FFDIM * DMODEL;

        // h = LN_in(rc_utt) -> bf16
        ln_kernel<<<lnBlocks, 256, 0, stream>>>(rcu, ln_in_w + l * DMODEL, ln_in_b + l * DMODEL,
                                                nullptr, h_bf, MTOK);
        // q = h @ Wq + bq ; kv = h @ Wkv + bkv
        gemm_bf16_kernel<false, true, false, false>
            <<<blocks(MTOK, DMODEL), 256, 0, stream>>>(h_bf, wq_l, bq + l * DMODEL,
                                                       nullptr, qbuf, nullptr,
                                                       MTOK, DMODEL, DMODEL);
        gemm_bf16_kernel<false, true, false, false>
            <<<blocks(MTOK, 2 * DMODEL), 256, 0, stream>>>(h_bf, wkv_l, bkv + l * 2 * DMODEL,
                                                           nullptr, kvbuf, nullptr,
                                                           MTOK, 2 * DMODEL, DMODEL);
        // masked attention (sparse window), bf16 out
        attn_kernel<<<(NROWS * BATCH * NH + 255) / 256, 256, 0, stream>>>(qbuf, kvbuf,
                                                                          lengths, a_bf);
        // rc_utt += attn @ Wo + bo   (residual fused, in-place)
        gemm_bf16_kernel<true, true, false, false>
            <<<blocks(MTOK, DMODEL), 256, 0, stream>>>(a_bf, wo_l, bo + l * DMODEL,
                                                       rcu, rcu, nullptr,
                                                       MTOK, DMODEL, DMODEL);
        // f = LN_ff(rc_utt) -> bf16
        ln_kernel<<<lnBlocks, 256, 0, stream>>>(rcu, ff_ln_w + l * DMODEL, ff_ln_b + l * DMODEL,
                                                nullptr, f_bf, MTOK);
        // ff1 = relu(f @ W1 + b1) -> bf16 ; tmp = rc_utt + ff1 @ W2 + b2
        gemm_bf16_kernel<false, false, true, true>
            <<<blocks(MTOK, FFDIM), 256, 0, stream>>>(f_bf, w1_l, b1 + l * FFDIM,
                                                      nullptr, nullptr, ff1_bf,
                                                      MTOK, FFDIM, DMODEL);
        gemm_bf16_kernel<true, true, false, false>
            <<<blocks(MTOK, DMODEL), 256, 0, stream>>>(ff1_bf, w2_l, b2 + l * DMODEL,
                                                       rcu, tmp, nullptr,
                                                       MTOK, DMODEL, FFDIM);
        // rc_utt = LN_out(tmp)
        ln_kernel<<<lnBlocks, 256, 0, stream>>>(tmp, ln_out_w + l * DMODEL, ln_out_b + l * DMODEL,
                                                rcu, nullptr, MTOK);
    }

    finalize_kernel<<<(ULEN * BATCH * DMODEL + BATCH + 255) / 256, 256, 0, stream>>>(rcu, lengths,
                                                                                    (float*)d_out);
}
